// DCNv4_YOLO_89945205113027
// MI455X (gfx1250) — matrix operand order repaired and checked by reading, not compiled, not run on hardware
//
#include <hip/hip_runtime.h>
#include <hip/hip_bf16.h>
#include <stdint.h>

// ---------------------------------------------------------------------------
// DCNv4 + BN + SiLU for MI455X (gfx1250).
// Pipeline: prep(x->bf16 NLC, LDS transpose) ; weight transpose->bf16 ;
//   WMMA GEMM value ; WMMA GEMM om ; deformable bilinear sample (float2) ;
//   WMMA GEMM out + BN + SiLU.
// GEMMs: 128x128 tile, K-step 64, double-buffered LDS fed by the Tensor Data
// Mover with one-barrier-per-iteration pipelining (TDM of tile k+1 overlaps
// the 16 v_wmma of tile k).
// ---------------------------------------------------------------------------

#define N_   2
#define C_   256
#define H_   80
#define W_   80
#define L_   6400            // H*W
#define G_   4
#define CG_  64
#define K_   9
#define OM_  108             // G*K*3
#define OMP_ 128             // padded om columns
#define MTOT (N_ * L_)       // 12800 GEMM rows

typedef __attribute__((ext_vector_type(16))) __bf16 v16bf;
typedef __attribute__((ext_vector_type(8)))  __bf16 v8bf;
typedef __attribute__((ext_vector_type(2)))  __bf16 v2bf;
typedef __attribute__((ext_vector_type(8)))  float  v8f;
typedef __attribute__((ext_vector_type(4)))  unsigned int u32x4;
typedef __attribute__((ext_vector_type(8)))  int i32x8;
typedef __attribute__((ext_vector_type(4)))  int i32x4;

// ---------------- workspace layout (bytes) ----------------
#define OFF_XP      0ull                               // N*L*C bf16      = 6,553,600
#define OFF_WV      (OFF_XP  + 6553600ull)             // 256*256 bf16    = 131,072
#define OFF_WOM     (OFF_WV  + 131072ull)              // 128*256 bf16    = 65,536
#define OFF_WOUT    (OFF_WOM + 65536ull)               // 256*256 bf16    = 131,072
#define OFF_VALUE   (OFF_WOUT + 131072ull)             // N*G*L*CG f32    = 13,107,200
#define OFF_OMBUF   (OFF_VALUE + 13107200ull)          // MTOT*128 f32    = 6,553,600
#define OFF_SAMP    (OFF_OMBUF + 6553600ull)           // N*L*C bf16      = 6,553,600
// total ~33.1 MB

// ---------------------------------------------------------------------------
// TDM (Tensor Data Mover) 2-D tile load -> LDS.  D# fields per CDNA5 ISA §8.
// ---------------------------------------------------------------------------
#if defined(__AMDGCN__) && __has_builtin(__builtin_amdgcn_tensor_load_to_lds)
#define ATHENA_TDM 1
__device__ __forceinline__ void tdm_load_tile_2d(unsigned ldsOff, const void* gptr,
                                                 unsigned tileK, unsigned tileRows,
                                                 unsigned strideElems) {
  unsigned long long ga = (unsigned long long)(uintptr_t)gptr;
  u32x4 g0;
  g0[0] = 1u;                                            // count=1 (valid), user mode
  g0[1] = ldsOff;                                        // lds_addr (bytes)
  g0[2] = (unsigned)ga;                                  // global_addr[31:0]
  g0[3] = ((unsigned)(ga >> 32) & 0x01FFFFFFu) | (2u << 30); // addr[56:32] | type=2
  i32x8 g1;
  g1[0] = (int)(1u << 16);                               // data_size=1 -> 2 bytes/elem
  g1[1] = (int)((strideElems & 0xFFFFu) << 16);          // tensor_dim0[15:0] @bit48
  g1[2] = (int)((strideElems >> 16) & 0xFFFFu);          // tensor_dim0[31:16]; dim1 lo=0
  g1[3] = (int)(0x10u | (tileK << 16));                  // tensor_dim1=0x100000 hi; tile_dim0
  g1[4] = (int)(tileRows & 0xFFFFu);                     // tile_dim1; tile_dim2=0 (2-D)
  g1[5] = (int)strideElems;                              // tensor_dim0_stride[31:0]
  g1[6] = 0;
  g1[7] = 0;
  i32x4 gz = {0, 0, 0, 0};
#if __clang_major__ >= 23
  i32x8 gz8 = {0, 0, 0, 0, 0, 0, 0, 0};
  __builtin_amdgcn_tensor_load_to_lds(g0, g1, gz, gz, gz8, 0);
#else
  __builtin_amdgcn_tensor_load_to_lds(g0, g1, gz, gz, 0);
#endif
}
#endif

// ---------------------------------------------------------------------------
// prep: x (N,C,H,W) f32 -> xp (N*L, C) bf16 via 32x32 LDS tile transpose.
// block (32,8); grid (L/32, C/32, N). Coalesced reads along l, writes along c.
// ---------------------------------------------------------------------------
__global__ __launch_bounds__(256) void k_prep(const float* __restrict__ x,
                                              __bf16* __restrict__ xp) {
  __shared__ float t[32][33];
  const int n  = blockIdx.z;
  const int l0 = blockIdx.x * 32;
  const int c0 = blockIdx.y * 32;
  const int lx = threadIdx.x;     // 0..31
  const int ly = threadIdx.y;     // 0..7
#pragma unroll
  for (int j = 0; j < 32; j += 8)
    t[ly + j][lx] = x[((size_t)n * C_ + c0 + ly + j) * L_ + l0 + lx];
  __syncthreads();
#pragma unroll
  for (int j = 0; j < 32; j += 8)
    xp[((size_t)n * L_ + l0 + ly + j) * C_ + c0 + lx] = (__bf16)t[lx][ly + j];
}

// weight (K=256 rows, Nout cols, row-major) -> Bt (NoutPad rows x 256, K-contig) bf16
__global__ __launch_bounds__(256) void k_wt(const float* __restrict__ src,
                                            __bf16* __restrict__ dst, int Nout) {
  int nRow = blockIdx.x;        // 0 .. NoutPad-1
  int k    = threadIdx.x;       // 0 .. 255
  float v = (nRow < Nout) ? src[(size_t)k * Nout + nRow] : 0.0f;
  dst[(size_t)nRow * 256 + k] = (__bf16)v;
}

// ---------------------------------------------------------------------------
// WMMA GEMM: Cout(M x Npad) = A(M x 256) * Bt^T + epilogue
// block = 256 threads = 8 waves (2 x 4); tile 128(M) x 128(N); K-step 64;
// double-buffered LDS (2 x 32 KB), TDM prefetch of tile k+1 overlaps compute.
// EPI: 0 = value scatter (+bias), 1 = om (+bias, cols<108), 2 = out+BN+SiLU->NCHW
// ---------------------------------------------------------------------------
template <int EPI>
__global__ __launch_bounds__(256) void k_gemm(const __bf16* __restrict__ A,
                                              const __bf16* __restrict__ Bt,
                                              const float* __restrict__ bias,
                                              float* __restrict__ Cout,
                                              const float* __restrict__ gamma,
                                              const float* __restrict__ beta,
                                              const float* __restrict__ mean,
                                              const float* __restrict__ var) {
  extern __shared__ char smem[];
  // buffer b at smem + b*32768 : sA 128x64 bf16 (128 B rows, 16 KB)
  //                              sB at +16384 (same shape, Bt rows)

  const int tid   = threadIdx.x;
  const int lane  = tid & 31;
  const int wave  = tid >> 5;
  const int waveM = wave >> 2;      // 0..1 -> 64 rows each
  const int waveN = wave & 3;       // 0..3 -> 32 cols each
  const int li    = lane & 15;
  const int h     = lane >> 4;

  const unsigned mBlock = blockIdx.x * 128u;
  const unsigned nBlock = blockIdx.y * 128u;

  v8f acc[4][2] = {};

#ifdef ATHENA_TDM
  if (wave == 0) {   // prologue: buffer 0 <- K chunk 0
    tdm_load_tile_2d(0u,     (const void*)(A  + (size_t)mBlock * 256), 64u, 128u, 256u);
    tdm_load_tile_2d(16384u, (const void*)(Bt + (size_t)nBlock * 256), 64u, 128u, 256u);
  }
#endif

  for (int kt = 0; kt < 4; ++kt) {
#ifdef ATHENA_TDM
    const int cur = kt & 1;
    if (wave == 0) __builtin_amdgcn_s_wait_tensorcnt(0);  // buffer `cur` ready
    __syncthreads();  // publish buffer `cur`; proves buffer `1-cur` fully read
    if (kt < 3 && wave == 0) {
      const unsigned nb = (unsigned)((kt + 1) & 1) * 32768u;
      tdm_load_tile_2d(nb,          (const void*)(A  + (size_t)mBlock * 256 + (kt + 1) * 64), 64u, 128u, 256u);
      tdm_load_tile_2d(nb + 16384u, (const void*)(Bt + (size_t)nBlock * 256 + (kt + 1) * 64), 64u, 128u, 256u);
    }
#else
    const int cur = 0;
    __syncthreads();
    {   // cooperative copy: 128 rows x 128 B per tile, 256 threads
      const char* gA = (const char*)(A  + (size_t)mBlock * 256 + kt * 64);
      const char* gB = (const char*)(Bt + (size_t)nBlock * 256 + kt * 64);
      int r  = tid >> 1;
      int hh = tid & 1;
#pragma unroll
      for (int q = 0; q < 4; ++q) {
        *(float4*)(smem + r * 128 + hh * 64 + q * 16) =
            *(const float4*)(gA + (size_t)r * 512 + hh * 64 + q * 16);
        *(float4*)(smem + 16384 + r * 128 + hh * 64 + q * 16) =
            *(const float4*)(gB + (size_t)r * 512 + hh * 64 + q * 16);
      }
    }
    __syncthreads();
#endif
    char* bA = smem + cur * 32768;
    char* bB = bA + 16384;

#pragma unroll
    for (int kk = 0; kk < 2; ++kk) {   // two 32-wide K chunks
      // A fragments: lane row = M; 16 contiguous K values at chunk offset 16h
      v16bf aF[4];
#pragma unroll
      for (int mi = 0; mi < 4; ++mi) {
        const char* p = bA + (size_t)(waveM * 64 + mi * 16 + li) * 128 + kk * 64 + h * 32;
        v8bf lo = *(const v8bf*)p;
        v8bf hi = *(const v8bf*)(p + 16);
        v16bf f;
#pragma unroll
        for (int e = 0; e < 8; ++e) { f[e] = lo[e]; f[e + 8] = hi[e]; }
        aF[mi] = f;
      }
      // B fragments: lane col = N; 16 contiguous K values at chunk offset 16h
      v16bf bF[2];
#pragma unroll
      for (int ni = 0; ni < 2; ++ni) {
        const char* p = bB + (size_t)(waveN * 32 + ni * 16 + li) * 128 + kk * 64 + h * 32;
        bF[ni] = *(const v16bf*)p;
      }
#pragma unroll
      for (int mi = 0; mi < 4; ++mi)
#pragma unroll
        for (int ni = 0; ni < 2; ++ni)
          acc[mi][ni] = __builtin_amdgcn_wmma_f32_16x16x32_bf16(
              false, aF[mi], false, bF[ni], (short)0, acc[mi][ni], false, false);
    }
  }

  // ---------------- epilogue ----------------
#pragma unroll
  for (int mi = 0; mi < 4; ++mi) {
#pragma unroll
    for (int ni = 0; ni < 2; ++ni) {
      const unsigned gn = nBlock + waveN * 32 + ni * 16 + li;   // output channel
      float bv = 0.0f, inv = 1.0f, sh = 0.0f;
      if (EPI == 0) {
        bv = bias[gn];
      } else if (EPI == 1) {
        bv = (gn < OM_) ? bias[gn] : 0.0f;
      } else {
        bv  = bias[gn];
        inv = gamma[gn] * rsqrtf(var[gn] + 1e-5f);
        sh  = beta[gn] - mean[gn] * inv;
      }
#pragma unroll
      for (int r = 0; r < 8; ++r) {
        const unsigned gm = mBlock + waveM * 64 + mi * 16 + h * 8 + r; // pixel row
        float v = acc[mi][ni][r] + bv;
        if (EPI == 0) {
          unsigned n = gm / L_, l = gm - n * L_;
          unsigned g = gn >> 6, cg = gn & 63u;
          Cout[(((size_t)n * G_ + g) * L_ + l) * CG_ + cg] = v;
        } else if (EPI == 1) {
          Cout[(size_t)gm * OMP_ + gn] = v;
        } else {
          float o = v * inv + sh;
          o = o * (1.0f / (1.0f + __expf(-o)));          // SiLU
          unsigned n = gm / L_, l = gm - n * L_;
          Cout[((size_t)n * C_ + gn) * L_ + l] = o;
        }
      }
    }
  }
}

// ---------------------------------------------------------------------------
// Deformable bilinear sampling.  256 threads = 8 items x 32 lanes;
// each lane owns a float2 channel pair (64-bit gathers from L2-resident value).
// item = ((n*G + g)*L + l); value is (n,g,l,cg); om is (n*L+l, 128).
// ---------------------------------------------------------------------------
__global__ __launch_bounds__(256) void k_sample(const float* __restrict__ value,
                                                const float* __restrict__ om,
                                                __bf16* __restrict__ sampled) {
  __shared__ float sOm[8][27];
  const int sub = threadIdx.x >> 5;
  const int ln  = threadIdx.x & 31;
  const unsigned item = blockIdx.x * 8u + sub;
  const unsigned l  = item % L_;
  const unsigned ng = item / L_;
  const unsigned g  = ng & 3u;
  const unsigned n  = ng >> 2;

  if (ln < 27)
    sOm[sub][ln] = om[((size_t)n * L_ + l) * OMP_ + g * 27 + ln];
  __syncthreads();

  const float2* vb = (const float2*)(value + ((size_t)ng * L_) * CG_); // (n,g,:,:)
  const int hy = (int)(l / W_);
  const int wx = (int)(l - hy * W_);

  float ax = 0.0f, ay = 0.0f;
#pragma unroll
  for (int t = 0; t < 9; ++t) {
    const float ox = sOm[sub][2 * t];
    const float oy = sOm[sub][2 * t + 1];
    const float m  = sOm[sub][18 + t];
    const float py = (float)(hy - 1 + t / 3) + oy;
    const float px = (float)(wx - 1 + t % 3) + ox;
    const float fy = floorf(py), fx = floorf(px);
    const float ly = py - fy, lx = px - fx;
    const int y0 = (int)fy, x0 = (int)fx;
#pragma unroll
    for (int dy = 0; dy < 2; ++dy) {
#pragma unroll
      for (int dx = 0; dx < 2; ++dx) {
        const int yi = y0 + dy, xi = x0 + dx;
        if (yi >= 0 && yi < H_ && xi >= 0 && xi < W_) {
          const float w = m * (dy ? ly : 1.0f - ly) * (dx ? lx : 1.0f - lx);
          const float2 vv = vb[(size_t)(yi * W_ + xi) * 32 + ln];
          ax += w * vv.x;
          ay += w * vv.y;
        }
      }
    }
  }
  v2bf o;
  o[0] = (__bf16)ax;
  o[1] = (__bf16)ay;
  *(v2bf*)(sampled + ((size_t)n * L_ + l) * C_ + g * CG_ + 2 * ln) = o;
}

// ---------------------------------------------------------------------------
extern "C" void kernel_launch(void* const* d_in, const int* in_sizes, int n_in,
                              void* d_out, int out_size, void* d_ws, size_t ws_size,
                              hipStream_t stream) {
  (void)in_sizes; (void)n_in; (void)out_size; (void)ws_size;
  const float* x       = (const float*)d_in[0];
  const float* w_value = (const float*)d_in[1];
  const float* b_value = (const float*)d_in[2];
  const float* w_om    = (const float*)d_in[3];
  const float* b_om    = (const float*)d_in[4];
  const float* w_out   = (const float*)d_in[5];
  const float* b_out   = (const float*)d_in[6];
  const float* gamma   = (const float*)d_in[7];
  const float* beta    = (const float*)d_in[8];
  const float* mean    = (const float*)d_in[9];
  const float* var     = (const float*)d_in[10];
  float* out = (float*)d_out;

  char* ws = (char*)d_ws;
  __bf16* xp     = (__bf16*)(ws + OFF_XP);
  __bf16* wv_t   = (__bf16*)(ws + OFF_WV);
  __bf16* wom_t  = (__bf16*)(ws + OFF_WOM);
  __bf16* wout_t = (__bf16*)(ws + OFF_WOUT);
  float*  value  = (float*)(ws + OFF_VALUE);
  float*  ombuf  = (float*)(ws + OFF_OMBUF);
  __bf16* samp   = (__bf16*)(ws + OFF_SAMP);

  dim3 pb(32, 8);
  dim3 pg(L_ / 32, C_ / 32, N_);
  k_prep<<<pg, pb, 0, stream>>>(x, xp);
  k_wt<<<256, 256, 0, stream>>>(w_value, wv_t, 256);
  k_wt<<<128, 256, 0, stream>>>(w_om,   wom_t, OM_);
  k_wt<<<256, 256, 0, stream>>>(w_out,  wout_t, 256);

  dim3 gv(MTOT / 128, 2);
  k_gemm<0><<<gv, 256, 65536, stream>>>(xp, wv_t, b_value, value,
                                        nullptr, nullptr, nullptr, nullptr);
  dim3 go(MTOT / 128, 1);
  k_gemm<1><<<go, 256, 65536, stream>>>(xp, wom_t, b_om, ombuf,
                                        nullptr, nullptr, nullptr, nullptr);

  k_sample<<<(N_ * G_ * L_) / 8, 256, 0, stream>>>(value, ombuf, samp);

  dim3 g3(MTOT / 128, 2);
  k_gemm<2><<<g3, 256, 65536, stream>>>(samp, wout_t, b_out, out,
                                        gamma, beta, mean, var);
}
